// AGNNConv_39041252720978
// MI455X (gfx1250) — matrix operand order repaired and checked by reading, not compiled, not run on hardware
//
#include <hip/hip_runtime.h>
#include <hip/hip_bf16.h>

#define N_NODES 50000
#define N_EDGES 800000
#define IN_DIM 256
#define OUT_DIM 64

typedef __attribute__((ext_vector_type(16))) __bf16 v16bf;
typedef __attribute__((ext_vector_type(8)))  float  v8f;

// Native fptrunc: backend selects hardware bf16 cvt (v_cvt_pk_bf16_f32) on
// gfx1250 instead of the 3-op software RNE sequence.
__device__ __forceinline__ __bf16 f2bf(float f) { return (__bf16)f; }

// ---------------------------------------------------------------------------
// Kernel 1: X_prime = X @ W using v_wmma_f32_16x16x32_bf16.
// One wave per 16-row tile; W staged in LDS (bf16, transposed, padded).
// ---------------------------------------------------------------------------
#define LDS_KPAD (IN_DIM + 8)   // pad row stride to kill LDS bank conflicts

__global__ __launch_bounds__(256) void agnn_gemm_wmma(
    const float* __restrict__ X, const float* __restrict__ W,
    float* __restrict__ Xp, int n_tiles)
{
    __shared__ __bf16 lw[OUT_DIM * LDS_KPAD];   // W^T in bf16: [n][k], ~33 KB

    const int tid = threadIdx.x;
    // cooperative stage of W (k-major in memory) -> transposed bf16 in LDS
    for (int idx = tid; idx < OUT_DIM * IN_DIM; idx += 256) {
        int n = idx & (OUT_DIM - 1);
        int k = idx >> 6;
        lw[n * LDS_KPAD + k] = f2bf(W[k * OUT_DIM + n]);
    }
    __syncthreads();

    const int wave = tid >> 5;
    const int lane = tid & 31;
    const int tile = blockIdx.x * 8 + wave;
    if (tile >= n_tiles) return;            // wave-uniform, no barriers follow

    const int half  = lane >> 4;            // 0: lanes 0-15, 1: lanes 16-31
    const int l16   = lane & 15;
    const int row   = tile * 16 + l16;      // A-matrix: lane = M row
    const float* xrow = X + (size_t)row * IN_DIM;

    // ---- load + convert all 8 K-step A fragments (K=256 = 8 x 32) ----
    // 16-bit A 16x32 layout: lanes 0-15 hold K 0-7 & 16-23; lanes 16-31 K 8-15 & 24-31
    v16bf afrag[8];
    const int khalf = half << 3;            // 0 or 8
#pragma unroll
    for (int kk = 0; kk < 8; ++kk) {
        const int base = kk * 32 + khalf;
        float4 lo0 = *(const float4*)(xrow + base);
        float4 lo1 = *(const float4*)(xrow + base + 4);
        float4 hi0 = *(const float4*)(xrow + base + 16);
        float4 hi1 = *(const float4*)(xrow + base + 20);
        afrag[kk][0]  = f2bf(lo0.x); afrag[kk][1]  = f2bf(lo0.y);
        afrag[kk][2]  = f2bf(lo0.z); afrag[kk][3]  = f2bf(lo0.w);
        afrag[kk][4]  = f2bf(lo1.x); afrag[kk][5]  = f2bf(lo1.y);
        afrag[kk][6]  = f2bf(lo1.z); afrag[kk][7]  = f2bf(lo1.w);
        afrag[kk][8]  = f2bf(hi0.x); afrag[kk][9]  = f2bf(hi0.y);
        afrag[kk][10] = f2bf(hi0.z); afrag[kk][11] = f2bf(hi0.w);
        afrag[kk][12] = f2bf(hi1.x); afrag[kk][13] = f2bf(hi1.y);
        afrag[kk][14] = f2bf(hi1.z); afrag[kk][15] = f2bf(hi1.w);
    }

    // ---- 4 output tiles of 16x16 across N=64 ----
    const int kboff = half << 3;            // B: lanes 0-15 K 0-7 & 16-23, lanes 16-31 K 8-15 & 24-31
#pragma unroll
    for (int nt = 0; nt < 4; ++nt) {
        v8f acc = {};
#pragma unroll
        for (int kk = 0; kk < 8; ++kk) {
            const __bf16* bp = &lw[(nt * 16 + l16) * LDS_KPAD + kk * 32 + kboff];
            v16bf bfrag;
#pragma unroll
            for (int e = 0; e < 16; ++e) bfrag[e] = bp[e < 8 ? e : e + 8];
            acc = __builtin_amdgcn_wmma_f32_16x16x32_bf16(
                false, afrag[kk], false, bfrag, (short)0, acc, false, false);
        }
        // C/D layout: VGPR i -> M=i (lanes 0-15) / M=8+i (lanes 16-31); N=lane&15
        const int mbase = tile * 16 + (half << 3);
#pragma unroll
        for (int i = 0; i < 8; ++i)
            Xp[(size_t)(mbase + i) * OUT_DIM + nt * 16 + l16] = acc[i];
    }
}

// ---------------------------------------------------------------------------
// Kernel 2: fused SDDMM + exp + segment-sum + normalize.
// edge_src is sorted -> one wave per node segment (binary search bounds).
// Lane owns 2 of the 64 output dims; per-edge dot via wave32 shfl_xor tree.
// Deterministic, atomic-free.
// ---------------------------------------------------------------------------
__global__ __launch_bounds__(256) void agnn_edge_agg(
    const float* __restrict__ Xp, const float* __restrict__ attw,
    const int* __restrict__ esrc, const int* __restrict__ edst,
    float* __restrict__ out, int n_nodes, int n_edges)
{
    const int wave = threadIdx.x >> 5;
    const int lane = threadIdx.x & 31;
    const int node = blockIdx.x * 8 + wave;
    if (node >= n_nodes) return;

    const float aw = attw[0];

    // source row fragment (resident)
    const float2 s = *(const float2*)(Xp + (size_t)node * OUT_DIM + 2 * lane);

    // lower_bound(esrc, node) and lower_bound(esrc, node+1), redundant per lane
    int lo = 0, hi = n_edges;
    while (lo < hi) { int mid = (lo + hi) >> 1; if (esrc[mid] <  node) lo = mid + 1; else hi = mid; }
    const int start = lo;
    hi = n_edges;
    while (lo < hi) { int mid = (lo + hi) >> 1; if (esrc[mid] <= node) lo = mid + 1; else hi = mid; }
    const int end = lo;

    float2 acc = make_float2(0.f, 0.f);
    float rowsum = 0.f;

    for (int e = start; e < end; ++e) {
        const int d = edst[e];
        if (e + 1 < end) {  // pull next dst row toward L2/L0 early
            __builtin_prefetch(Xp + (size_t)edst[e + 1] * OUT_DIM + 2 * lane, 0, 3);
        }
        const float2 t = *(const float2*)(Xp + (size_t)d * OUT_DIM + 2 * lane);
        // per-edge dot over 64 dims: 2 per lane + wave32 reduction
        float p = s.x * t.x + s.y * t.y;
#pragma unroll
        for (int off = 16; off > 0; off >>= 1) p += __shfl_xor(p, off, 32);
        const float att = __expf(p * aw);
        acc.x += att * t.x;
        acc.y += att * t.y;
        rowsum += att;
    }

    // h_prime = agg / rows_sum  (0/0 -> NaN matches reference semantics)
    out[(size_t)node * OUT_DIM + 2 * lane]     = acc.x / rowsum;
    out[(size_t)node * OUT_DIM + 2 * lane + 1] = acc.y / rowsum;
}

// ---------------------------------------------------------------------------
extern "C" void kernel_launch(void* const* d_in, const int* in_sizes, int n_in,
                              void* d_out, int out_size, void* d_ws, size_t ws_size,
                              hipStream_t stream) {
    const float* X    = (const float*)d_in[0];   // [N_NODES, IN_DIM]
    const float* W    = (const float*)d_in[1];   // [IN_DIM, OUT_DIM]
    const float* attw = (const float*)d_in[2];   // [1]
    const int*   esrc = (const int*)d_in[3];     // [N_EDGES] sorted
    const int*   edst = (const int*)d_in[4];     // [N_EDGES]
    float*       out  = (float*)d_out;           // [N_NODES, OUT_DIM]
    float*       Xp   = (float*)d_ws;            // scratch: [N_NODES, OUT_DIM] = 12.8 MB

    (void)in_sizes; (void)n_in; (void)out_size; (void)ws_size;

    const int n_tiles = N_NODES / 16;            // 50000 % 16 == 0 -> 3125
    agnn_gemm_wmma<<<(n_tiles + 7) / 8, 256, 0, stream>>>(X, W, Xp, n_tiles);

    agnn_edge_agg<<<(N_NODES + 7) / 8, 256, 0, stream>>>(
        Xp, attw, esrc, edst, out, N_NODES, N_EDGES);
}
